// get_model_47510928228929
// MI455X (gfx1250) — matrix operand order repaired and checked by reading, not compiled, not run on hardware
//
#include <hip/hip_runtime.h>
#include <hip/hip_bf16.h>

// ---------------------------------------------------------------------------
// PointNet++ MSG inference for MI455X (gfx1250, wave32, WMMA).
// Pipeline: FPS (shared) -> ball query (shared) -> per-branch weight
// pre-conversion to padded bf16 -> fused gather+MLP+maxpool per centroid
// using v_wmma_f32_16x16x32_bf16 -> FC head.
// ---------------------------------------------------------------------------

typedef __attribute__((ext_vector_type(16))) __bf16          v16bf;
typedef __attribute__((ext_vector_type(8)))  float           v8f;
typedef __attribute__((ext_vector_type(8)))  unsigned short  us8;

union BF16Frag { v16bf v; us8 h[2]; unsigned short u[16]; };

__device__ __forceinline__ unsigned short f2bf(float f) {
  unsigned int x = __float_as_uint(f);
  return (unsigned short)((x + 0x7fffu + ((x >> 16) & 1u)) >> 16);  // RNE
}

// ---------------------------- Farthest point sampling ----------------------
// One block per batch; distances live in registers (<=16 pts/thread).
__global__ void __launch_bounds__(256) fps_kernel(
    const float* __restrict__ crd, long long cBS, long long cCS, long long cPS,
    int N, int npoint, int* __restrict__ outIdx)
{
  __shared__ float sval[256];
  __shared__ int   sidx[256];
  __shared__ int   sfar;
  const int tid = threadIdx.x;
  const int b = blockIdx.x;
  const int nPer = (N + 255) >> 8;
  float dist[16];
  for (int i = 0; i < nPer; ++i) dist[i] = 1e10f;
  int far = 0;
  for (int it = 0; it < npoint; ++it) {
    if (tid == 0) outIdx[b * npoint + it] = far;
    const float cx = crd[b*cBS + 0*cCS + (long long)far*cPS];
    const float cy = crd[b*cBS + 1*cCS + (long long)far*cPS];
    const float cz = crd[b*cBS + 2*cCS + (long long)far*cPS];
    float best = -1.0f; int bidx = 0;
    for (int i = 0; i < nPer; ++i) {
      int p = tid + (i << 8);
      if (p < N) {
        float dx = crd[b*cBS + 0*cCS + (long long)p*cPS] - cx;
        float dy = crd[b*cBS + 1*cCS + (long long)p*cPS] - cy;
        float dz = crd[b*cBS + 2*cCS + (long long)p*cPS] - cz;
        float d = dx*dx + dy*dy + dz*dz;
        float dd = dist[i] < d ? dist[i] : d;
        dist[i] = dd;
        if (dd > best) { best = dd; bidx = p; }
      }
    }
    sval[tid] = best; sidx[tid] = bidx;
    __syncthreads();
    for (int off = 128; off > 0; off >>= 1) {
      if (tid < off) {
        float v2 = sval[tid + off]; int i2 = sidx[tid + off];
        if (v2 > sval[tid] || (v2 == sval[tid] && i2 < sidx[tid])) {
          sval[tid] = v2; sidx[tid] = i2;
        }
      }
      __syncthreads();
    }
    if (tid == 0) sfar = sidx[0];
    __syncthreads();
    far = sfar;
  }
}

__global__ void gather_xyz_kernel(
    const float* __restrict__ crd, long long cBS, long long cCS, long long cPS,
    const int* __restrict__ fpsIdx, int S, int total, float* __restrict__ outXYZ)
{
  int t = blockIdx.x * blockDim.x + threadIdx.x;
  if (t >= total) return;
  int c = t % 3; int s = (t / 3) % S; int b = t / (3 * S);
  int p = fpsIdx[b * S + s];
  outXYZ[t] = crd[b*cBS + c*cCS + (long long)p*cPS];
}

// ---------------------------- Ball query -----------------------------------
__global__ void ball_query_kernel(
    const float* __restrict__ crd, long long cBS, long long cCS, long long cPS,
    const float* __restrict__ newxyz, int N, int S, int total,
    float r2, int K, int* __restrict__ outIdx)
{
  int t = blockIdx.x * blockDim.x + threadIdx.x;
  if (t >= total) return;
  int b = t / S;
  const float cx = newxyz[(size_t)t*3+0];
  const float cy = newxyz[(size_t)t*3+1];
  const float cz = newxyz[(size_t)t*3+2];
  int* dst = outIdx + (size_t)t * K;
  int cnt = 0; int first = 0;
  for (int p = 0; p < N && cnt < K; ++p) {
    float dx = crd[b*cBS + 0*cCS + (long long)p*cPS] - cx;
    float dy = crd[b*cBS + 1*cCS + (long long)p*cPS] - cy;
    float dz = crd[b*cBS + 2*cCS + (long long)p*cPS] - cz;
    if (dx*dx + dy*dy + dz*dz <= r2) {
      if (cnt == 0) first = p;
      dst[cnt++] = p;
    }
  }
  for (; cnt < K; ++cnt) dst[cnt] = first;
}

// ------------------ Weight conversion: fp32 [Cout,Cin] -> bf16 [Cout,Cpad] --
// Pad region zero-filled so WMMA B fragments need no bounds checks.
__global__ void wcvt_kernel(const float* __restrict__ W, unsigned short* __restrict__ WB,
                            int total, int Cin, int Cpad)
{
  int t = blockIdx.x * blockDim.x + threadIdx.x;
  if (t >= total) return;
  int n = t / Cpad, k = t - n * Cpad;
  WB[t] = (k < Cin) ? f2bf(W[(size_t)n * Cin + k]) : (unsigned short)0;
}

// -------------------- Fused gather + 3-layer MLP + max-pool -----------------
// One workgroup (8 wave32) per (batch, centroid). Activations in LDS as bf16;
// weights are pre-padded bf16 in global (L2-resident): B fragment = two
// contiguous 16B loads per lane. Last layer max-pooled on the fly.
__global__ void __launch_bounds__(256) fused_group_mlp(
    const float* __restrict__ feat, long long fBS, long long fCS,
    const float* __restrict__ crd, long long cBS, long long cCS, long long cPS,
    const float* __restrict__ newxyz,
    const int* __restrict__ gidx,
    const unsigned short* __restrict__ wb1, const float* __restrict__ bz1,
    const unsigned short* __restrict__ wb2, const float* __restrict__ bz2,
    const unsigned short* __restrict__ wb3, const float* __restrict__ bz3,
    int Cin, int Cp, int C1, int C2, int C3,
    int S, int K, int xyzFirst,
    float* __restrict__ out, int outTotal, int chOff)
{
  extern __shared__ char smem[];
  const int tid  = threadIdx.x;
  const int lane = tid & 31;
  const int wave = tid >> 5;
  const int nW   = blockDim.x >> 5;
  const int b = blockIdx.x / S;
  const int s = blockIdx.x % S;

  const int Cp0 = (Cin + 31) & ~31;
  const int Cp1 = (C1  + 31) & ~31;   // == C1 (all hidden widths are mult of 32)
  const int Cp2 = (C2  + 31) & ~31;   // == C2
  const size_t h0 = (size_t)K * Cp0 * 2;
  const size_t h1 = (size_t)K * Cp1 * 2;
  const size_t h2 = (size_t)K * Cp2 * 2;
  unsigned short* H0 = (unsigned short*)smem;
  unsigned short* H1 = (unsigned short*)(smem + h0);
  unsigned short* H2 = (h2 <= h0) ? H0 : (unsigned short*)(smem + h0 + h1);
  unsigned int*  smax = (unsigned int*)(smem + h0 + h1 + ((h2 <= h0) ? 0 : h2));

  // gather grouped features (f32 -> bf16), zero-pad Cin -> Cp0
  for (int t = tid; t < K * Cp0; t += blockDim.x) {
    int j = t / Cp0, c = t - j * Cp0;
    float v = 0.f;
    if (c < Cin) {
      int p = gidx ? gidx[(size_t)(b * S + s) * K + j] : j;
      if (xyzFirst) {                         // group_all: [xyz, feat], no centering
        v = (c < 3) ? crd[b*cBS + c*cCS + (long long)p*cPS]
                    : feat[b*fBS + (long long)(c - 3)*fCS + p];
      } else if (c < Cp) {                    // MSG: [feat, xyz - centroid]
        v = feat[b*fBS + (long long)c*fCS + p];
      } else {
        int cc = c - Cp;
        v = crd[b*cBS + cc*cCS + (long long)p*cPS] - newxyz[(size_t)(b*S+s)*3 + cc];
      }
    }
    H0[t] = f2bf(v);
  }
  for (int t = tid; t < C3; t += blockDim.x) smax[t] = 0u;
  __syncthreads();

  const unsigned short* Ain[3]  = { H0, H1, H2 };
  unsigned short*       Aout[3] = { H1, H2, nullptr };
  const unsigned short* W[3]  = { wb1, wb2, wb3 };
  const float*          BI[3] = { bz1, bz2, bz3 };
  const int cpadL[3] = { Cp0, Cp1, Cp2 };
  const int coutL[3] = { C1, C2, C3 };

  const int Mt   = K >> 4;
  const int half = lane >> 4;
  const int l16  = lane & 15;

  #pragma unroll
  for (int l = 0; l < 3; ++l) {
    const int Nt   = coutL[l] >> 4;
    const int Kt   = cpadL[l] >> 5;
    const int cpad = cpadL[l];
    const unsigned short* A  = Ain[l];
    const unsigned short* Wl = W[l];
    for (int t = wave; t < Mt * Nt; t += nW) {
      const int mt = t % Mt, nt = t / Mt;
      const int n   = nt * 16 + l16;
      const int row = mt * 16 + l16;
      const float bv = BI[l][n];
      v8f acc;
      #pragma unroll
      for (int j = 0; j < 8; ++j) acc[j] = bv;
      const unsigned short* Arow = A  + (size_t)row * cpad;
      const unsigned short* Wrow = Wl + (size_t)n   * cpad;
      for (int kt = 0; kt < Kt; ++kt) {
        BF16Frag af, bf;
        // A 16x32 bf16 layout: lanes<16 hold K 0-7 / 16-23, lanes>=16 hold 8-15 / 24-31
        const int ka = kt * 32 + half * 8;
        af.h[0] = *(const us8*)(Arow + ka);        // ds_load_b128
        af.h[1] = *(const us8*)(Arow + ka + 16);   // ds_load_b128
        // B 32x16 layout: lane = column, lanes<16 hold K 0-7 / 16-23, lanes>=16 hold 8-15 / 24-31
        const int kb = kt * 32 + half * 8;
        bf.h[0] = *(const us8*)(Wrow + kb);        // global_load_b128
        bf.h[1] = *(const us8*)(Wrow + kb + 16);   // global_load_b128
        acc = __builtin_amdgcn_wmma_f32_16x16x32_bf16(false, af.v, false, bf.v,
                                                      (short)0, acc, false, false);
      }
      if (l < 2) {
        unsigned short* HO = Aout[l];
        const int co = coutL[l];            // next layer's cpad == cout
        #pragma unroll
        for (int j = 0; j < 8; ++j) {
          float v = acc[j] > 0.f ? acc[j] : 0.f;
          int m = mt * 16 + j + (half << 3);  // D layout: vgpr j -> row j / j+8
          HO[(size_t)m * co + n] = f2bf(v);
        }
      } else {
        float mx = 0.f;                     // fused max-pool over group dim
        #pragma unroll
        for (int j = 0; j < 8; ++j) {
          float v = acc[j] > 0.f ? acc[j] : 0.f;
          mx = mx > v ? mx : v;
        }
        atomicMax(&smax[n], __float_as_uint(mx));  // relu>=0: uint order == float order
      }
    }
    __syncthreads();
  }

  for (int c = tid; c < C3; c += blockDim.x)
    out[((size_t)b * outTotal + chOff + c) * S + s] = __uint_as_float(smax[c]);
}

// ---------------------------- Dense FC head ---------------------------------
__global__ void fc_kernel(const float* __restrict__ x, const float* __restrict__ w,
                          const float* __restrict__ bias, float* __restrict__ y,
                          int total, int Cin, int Cout, int act)
{
  int t = blockIdx.x * blockDim.x + threadIdx.x;
  if (t >= total) return;
  int b = t / Cout, o = t % Cout;
  const float* xr = x + (size_t)b * Cin;
  const float* wr = w + (size_t)o * Cin;
  float acc = bias[o];
  for (int c = 0; c < Cin; ++c) acc += xr[c] * wr[c];
  if (act == 0)      acc = acc > 0.f ? acc : 0.f;
  else if (act == 1) acc = 1.f / (1.f + __expf(-acc));
  y[t] = acc;
}

// ---------------------------------------------------------------------------
extern "C" void kernel_launch(void* const* d_in, const int* in_sizes, int n_in,
                              void* d_out, int out_size, void* d_ws, size_t ws_size,
                              hipStream_t stream)
{
  (void)in_sizes; (void)n_in; (void)out_size; (void)ws_size;
  const float* xyz = (const float*)d_in[0];

  struct Conv { const float* w; const float* b; };
  Conv sa1[3][3][3], sa2[3][3][3], sa3[3][3], fc[3];
  int pi = 1;  // flatten order: b1{sa1[3][3],sa2[3][3],sa3[3]}, b2, b3, fc1..3
  for (int br = 0; br < 3; ++br) {
    for (int sc = 0; sc < 3; ++sc)
      for (int l = 0; l < 3; ++l) { sa1[br][sc][l].w = (const float*)d_in[pi++]; sa1[br][sc][l].b = (const float*)d_in[pi++]; }
    for (int sc = 0; sc < 3; ++sc)
      for (int l = 0; l < 3; ++l) { sa2[br][sc][l].w = (const float*)d_in[pi++]; sa2[br][sc][l].b = (const float*)d_in[pi++]; }
    for (int l = 0; l < 3; ++l)   { sa3[br][l].w = (const float*)d_in[pi++]; sa3[br][l].b = (const float*)d_in[pi++]; }
  }
  for (int i = 0; i < 3; ++i) { fc[i].w = (const float*)d_in[pi++]; fc[i].b = (const float*)d_in[pi++]; }

  const int Bn = 8, N = 4096, S1 = 512, S2 = 128;
  const float R1[3] = {0.1f, 0.2f, 0.4f}; const int K1[3] = {16, 32, 128};
  const int M1[3][4] = {{6,32,32,64},{6,64,64,128},{6,64,96,128}};
  const int OFF1[3] = {0, 64, 192}; const int CT1 = 320;
  const float R2[3] = {0.2f, 0.4f, 0.8f}; const int K2[3] = {32, 64, 128};
  const int M2[3][4] = {{323,64,64,128},{323,128,128,256},{323,128,128,256}};
  const int OFF2[3] = {0, 128, 384}; const int CT2 = 640;
  const int M3[4] = {643, 256, 512, 1024};

  // workspace carve-out (~14 MB total; branch buffers reused serially)
  char* ws = (char*)d_ws;
  size_t off = 0;
  auto carve = [&](size_t bytes) { char* p = ws + off; off = (off + bytes + 255) & ~(size_t)255; return p; };
  int*   fps1 = (int*)carve((size_t)Bn * S1 * 4);
  float* nx1  = (float*)carve((size_t)Bn * S1 * 3 * 4);
  int*   fps2 = (int*)carve((size_t)Bn * S2 * 4);
  float* nx2  = (float*)carve((size_t)Bn * S2 * 3 * 4);
  int* idx1[3]; for (int r = 0; r < 3; ++r) idx1[r] = (int*)carve((size_t)Bn * S1 * K1[r] * 4);
  int* idx2[3]; for (int r = 0; r < 3; ++r) idx2[r] = (int*)carve((size_t)Bn * S2 * K2[r] * 4);
  float* l1p  = (float*)carve((size_t)Bn * CT1 * S1 * 4);
  float* l2p  = (float*)carve((size_t)Bn * CT2 * S2 * 4);
  float* fcin = (float*)carve((size_t)Bn * 3072 * 4);
  float* h1   = (float*)carve((size_t)Bn * 1024 * 4);
  float* h2   = (float*)carve((size_t)Bn * 256 * 4);

  // per-branch padded-bf16 weight scratch (21 conv layers, identical shapes
  // across branches -> fixed offset table, region reused per branch)
  struct LayerDesc { int cin, cout, cpad; size_t off; };
  LayerDesc ld[21]; int nl = 0; size_t woff = 0;
  auto addL = [&](int cin, int cout) {
    int cp = (cin + 31) & ~31;
    ld[nl].cin = cin; ld[nl].cout = cout; ld[nl].cpad = cp; ld[nl].off = woff;
    woff += (size_t)cout * cp * 2; woff = (woff + 255) & ~(size_t)255; ++nl;
  };
  for (int sc = 0; sc < 3; ++sc) for (int l = 0; l < 3; ++l) addL(M1[sc][l], M1[sc][l+1]);
  for (int sc = 0; sc < 3; ++sc) for (int l = 0; l < 3; ++l) addL(M2[sc][l], M2[sc][l+1]);
  for (int l = 0; l < 3; ++l) addL(M3[l], M3[l+1]);
  unsigned short* wbuf = (unsigned short*)carve(woff);
  auto WB = [&](int i) { return wbuf + ld[i].off / 2; };

  const long long xBS = 6LL*N, xCS = N, xPS = 1;      // xyz  [B,6,N]
  const long long n1BS = 3LL*S1, n1CS = 1, n1PS = 3;  // nx1  [B,S1,3]
  const long long n2BS = 3LL*S2, n2CS = 1, n2PS = 3;  // nx2  [B,S2,3]

  // 1) FPS + centroid gather (coords identical across branches -> once)
  fps_kernel<<<Bn, 256, 0, stream>>>(xyz, xBS, xCS, xPS, N, S1, fps1);
  { int tot = Bn*S1*3; gather_xyz_kernel<<<(tot+255)/256, 256, 0, stream>>>(xyz, xBS, xCS, xPS, fps1, S1, tot, nx1); }
  fps_kernel<<<Bn, 256, 0, stream>>>(nx1, n1BS, n1CS, n1PS, S1, S2, fps2);
  { int tot = Bn*S2*3; gather_xyz_kernel<<<(tot+255)/256, 256, 0, stream>>>(nx1, n1BS, n1CS, n1PS, fps2, S2, tot, nx2); }

  // 2) ball queries (also branch-independent)
  for (int r = 0; r < 3; ++r) {
    int tot = Bn * S1;
    ball_query_kernel<<<(tot+255)/256, 256, 0, stream>>>(xyz, xBS, xCS, xPS, nx1, N, S1, tot,
                                                         R1[r]*R1[r], K1[r], idx1[r]);
  }
  for (int r = 0; r < 3; ++r) {
    int tot = Bn * S2;
    ball_query_kernel<<<(tot+255)/256, 256, 0, stream>>>(nx1, n1BS, n1CS, n1PS, nx2, S1, S2, tot,
                                                         R2[r]*R2[r], K2[r], idx2[r]);
  }

  auto shmemSize = [](int K, int Cin, int C1, int C2, int C3) -> size_t {
    int cp0 = (Cin+31)&~31, cp1 = (C1+31)&~31, cp2 = (C2+31)&~31;
    size_t h0 = (size_t)K*cp0*2, h1 = (size_t)K*cp1*2, h2 = (size_t)K*cp2*2;
    return h0 + h1 + ((h2 <= h0) ? 0 : h2) + (size_t)C3*4;
  };

  // 3) per-branch: convert weights to padded bf16, then SA stages
  for (int br = 0; br < 3; ++br) {
    const Conv* seq[21]; int li = 0;
    for (int sc = 0; sc < 3; ++sc) for (int l = 0; l < 3; ++l) seq[li++] = &sa1[br][sc][l];
    for (int sc = 0; sc < 3; ++sc) for (int l = 0; l < 3; ++l) seq[li++] = &sa2[br][sc][l];
    for (int l = 0; l < 3; ++l) seq[li++] = &sa3[br][l];
    for (int i = 0; i < 21; ++i) {
      int tot = ld[i].cout * ld[i].cpad;
      wcvt_kernel<<<(tot+255)/256, 256, 0, stream>>>(seq[i]->w, WB(i), tot, ld[i].cin, ld[i].cpad);
    }

    const float* norm = xyz + (size_t)(3 + br) * N;  // channel i repeated x3 => fCS=0
    for (int sc = 0; sc < 3; ++sc) {
      const int* M = M1[sc];
      const int L = sc * 3;
      fused_group_mlp<<<Bn*S1, 256, shmemSize(K1[sc], M[0], M[1], M[2], M[3]), stream>>>(
        norm, 6LL*N, 0, xyz, xBS, xCS, xPS, nx1, idx1[sc],
        WB(L+0), sa1[br][sc][0].b, WB(L+1), sa1[br][sc][1].b, WB(L+2), sa1[br][sc][2].b,
        M[0], 3, M[1], M[2], M[3], S1, K1[sc], 0, l1p, CT1, OFF1[sc]);
    }
    for (int sc = 0; sc < 3; ++sc) {
      const int* M = M2[sc];
      const int L = 9 + sc * 3;
      fused_group_mlp<<<Bn*S2, 256, shmemSize(K2[sc], M[0], M[1], M[2], M[3]), stream>>>(
        l1p, (long long)CT1*S1, S1, nx1, n1BS, n1CS, n1PS, nx2, idx2[sc],
        WB(L+0), sa2[br][sc][0].b, WB(L+1), sa2[br][sc][1].b, WB(L+2), sa2[br][sc][2].b,
        M[0], CT1, M[1], M[2], M[3], S2, K2[sc], 0, l2p, CT2, OFF2[sc]);
    }
    fused_group_mlp<<<Bn, 256, shmemSize(S2, M3[0], M3[1], M3[2], M3[3]), stream>>>(
      l2p, (long long)CT2*S2, S2, nx2, n2BS, n2CS, n2PS, nullptr, nullptr,
      WB(18), sa3[br][0].b, WB(19), sa3[br][1].b, WB(20), sa3[br][2].b,
      M3[0], CT2, M3[1], M3[2], M3[3], 1, S2, 1, fcin, 3072, br * 1024);
  }

  // 4) FC head (tiny: 8 rows)
  { int tot = Bn*1024; fc_kernel<<<(tot+255)/256, 256, 0, stream>>>(fcin, fc[0].w, fc[0].b, h1, tot, 3072, 1024, 0); }
  { int tot = Bn*256;  fc_kernel<<<(tot+255)/256, 256, 0, stream>>>(h1,   fc[1].w, fc[1].b, h2, tot, 1024, 256, 0); }
  { int tot = Bn*1;    fc_kernel<<<(tot+255)/256, 256, 0, stream>>>(h2,   fc[2].w, fc[2].b, (float*)d_out, tot, 256, 1, 1); }
}
